// NMTCGRUEncoder_30958124269569
// MI455X (gfx1250) — matrix operand order repaired and checked by reading, not compiled, not run on hardware
//
#include <hip/hip_runtime.h>
#include <hip/hip_bf16.h>

// Bidirectional GRU encoder for MI455X (gfx1250), bf16 WMMA path.
// V=32000 E=512 H=1024 B=64 S=256.

typedef __bf16 bf16_t;
typedef __attribute__((ext_vector_type(16))) __bf16 v16bf;
typedef __attribute__((ext_vector_type(8)))  __bf16 v8bf;
typedef __attribute__((ext_vector_type(8)))  float  v8f;

#define V_ 32000
#define E_ 512
#define H_ 1024
#define B_ 64
#define S_ 256
#define NT_ 192          // 3H/16 output col tiles per direction

#define WMMA_BF16(a, b, c) \
  __builtin_amdgcn_wmma_f32_16x16x32_bf16(false, (a), false, (b), (short)0, (c), false, false)

__device__ __forceinline__ bf16_t f2bf(float f) {
  union { float f; unsigned u; } x; x.f = f;
  unsigned r = x.u + 0x7FFFu + ((x.u >> 16) & 1u);   // round-to-nearest-even
  unsigned short h = (unsigned short)(r >> 16);
  bf16_t o; __builtin_memcpy(&o, &h, sizeof(o));
  return o;
}

// A operand (16x32, MxK) as loaded here (contiguous, matches B pack): slots 0..15 -> K = half*16 + s.
// Source row already bf16: two pure
// 16-byte loads, no conversion VALU in the hot loop.
__device__ __forceinline__ v16bf load_a_bf(const bf16_t* __restrict__ p, int half) {
  v8bf lo = *(const v8bf*)(p + half * 16);
  v8bf hi = *(const v8bf*)(p + half * 16 + 8);
  return __builtin_shufflevector(lo, hi, 0, 1, 2, 3, 4, 5, 6, 7,
                                 8, 9, 10, 11, 12, 13, 14, 15);
}

// Pack w_ih (3H x E, f32 row-major) -> bf16 B-operand tiles:
// flat [d][nt][kt][lane][slot], value = w[nt*16 + lane%16][kt*32 + (lane/16)*16 + slot].
__global__ __launch_bounds__(256) void pack_wih_kernel(
    const float* __restrict__ wf, const float* __restrict__ wr, bf16_t* __restrict__ dst) {
  int i = blockIdx.x * 256 + threadIdx.x;            // 2 * 192*16*512 = 3,145,728
  const int PER = NT_ * 16 * 512;
  int d = i / PER;  int r = i - d * PER;
  int nt = r >> 13; int r2 = r & 8191;               // 16*512 = 8192
  int kt = r2 >> 9; int r3 = r2 & 511;
  int lane = r3 >> 4, slot = r3 & 15;
  int g = nt * 16 + (lane & 15);
  int e = (kt << 5) + ((lane >> 4) << 4) + slot;
  const float* src = d ? wr : wf;
  dst[i] = f2bf(src[(size_t)g * E_ + e]);
}

// Same for w_hh (3H x H): flat [d][nt][kt(0..31)][lane][slot].
__global__ __launch_bounds__(256) void pack_whh_kernel(
    const float* __restrict__ wf, const float* __restrict__ wr, bf16_t* __restrict__ dst) {
  int i = blockIdx.x * 256 + threadIdx.x;            // 2 * 192*32*512 = 6,291,456
  const int PER = NT_ * 32 * 512;
  int d = i / PER;  int r = i - d * PER;
  int nt = r >> 14; int r2 = r & 16383;              // 32*512 = 16384
  int kt = r2 >> 9; int r3 = r2 & 511;
  int lane = r3 >> 4, slot = r3 & 15;
  int g = nt * 16 + (lane & 15);
  int e = (kt << 5) + ((lane >> 4) << 4) + slot;
  const float* src = d ? wr : wf;
  dst[i] = f2bf(src[(size_t)g * H_ + e]);
}

// Gather step-ordered bf16 inputs once: x_bf[d][t][b][e] = bf16(emb[token(d,t,b)][e]).
// For d==1 the time index is pre-reversed via the involution f(t)=S-1-((t+pad)%S).
__global__ __launch_bounds__(256) void gather_x_kernel(
    const int* __restrict__ x_source, const int* __restrict__ x_lengths,
    const float* __restrict__ emb, bf16_t* __restrict__ x_bf) {
  int i = blockIdx.x * 256 + threadIdx.x;            // 2*256*64*512 = 16,777,216
  int e = i & 511;
  int b = (i >> 9) & 63;
  int t = (i >> 15) & 255;
  int d = i >> 23;
  int pos = t;
  if (d == 1) {
    int len = x_lengths[b];
    pos = (S_ - 1) - ((t + (S_ - len)) & (S_ - 1));
  }
  int tok = x_source[b * S_ + pos];
  x_bf[i] = f2bf(emb[(size_t)tok * E_ + e]);
}

__global__ __launch_bounds__(256) void zero_state_kernel(
    float* __restrict__ state_f, bf16_t* __restrict__ state_bf) {
  int i = blockIdx.x * 256 + threadIdx.x;            // 4 * 64*1024 = 262,144
  state_f[i] = 0.0f;
  state_bf[i] = f2bf(0.0f);
}

// One GRU time step, both directions fused. One wave carries ALL FOUR batch
// tiles (M=64) for one hidden col tile: per K-chunk, B tiles loaded once and
// reused by 4 A rows -> 12 WMMAs per 14 x b128 loads, 4x less L2 weight
// traffic than one-tile-per-wave. 576 bf16 WMMAs per wave per step.
__global__ __launch_bounds__(256) void gru_step_kernel(
    const int* __restrict__ x_lengths,
    const bf16_t* __restrict__ x_bf,
    const bf16_t* __restrict__ wih, const bf16_t* __restrict__ whh,
    const float* __restrict__ b_ih_f, const float* __restrict__ b_hh_f,
    const float* __restrict__ b_ih_r, const float* __restrict__ b_hh_r,
    float* __restrict__ state_f, bf16_t* __restrict__ state_bf,
    float* __restrict__ out, int t) {
  const int lane = threadIdx.x & 31;
  const int wave = threadIdx.x >> 5;
  const int gw   = blockIdx.x * 8 + wave;   // 0..127
  const int d    = gw >> 6;                 // direction 0=fwd 1=rev
  const int ct   = gw & 63;                 // hidden col tile 0..63
  const int half = lane >> 4;
  const int nIdx = lane & 15;

  float*  h_prev_f  = state_f  + (size_t)(d * 2 + (t & 1)) * (B_ * H_);
  float*  h_next_f  = state_f  + (size_t)(d * 2 + ((t + 1) & 1)) * (B_ * H_);
  bf16_t* h_prev_bf = state_bf + (size_t)(d * 2 + (t & 1)) * (B_ * H_);
  bf16_t* h_next_bf = state_bf + (size_t)(d * 2 + ((t + 1) & 1)) * (B_ * H_);

  // A-operand rows: one per batch tile (same logical row in both lane-halves).
  const bf16_t* hrow[4];
  const bf16_t* xrow[4];
#pragma unroll
  for (int m = 0; m < 4; ++m) {
    int arow = m * 16 + nIdx;               // batch index
    hrow[m] = h_prev_bf + (size_t)arow * H_;
    xrow[m] = x_bf + ((size_t)(d * S_ + t) * B_ + arow) * E_;
  }

  const bf16_t* Bh_r = whh + ((size_t)(d * NT_ + ct      ) * 32) * 512 + lane * 16;
  const bf16_t* Bh_z = whh + ((size_t)(d * NT_ + 64 + ct ) * 32) * 512 + lane * 16;
  const bf16_t* Bh_n = whh + ((size_t)(d * NT_ + 128 + ct) * 32) * 512 + lane * 16;
  const bf16_t* Bx_r = wih + ((size_t)(d * NT_ + ct      ) * 16) * 512 + lane * 16;
  const bf16_t* Bx_z = wih + ((size_t)(d * NT_ + 64 + ct ) * 16) * 512 + lane * 16;
  const bf16_t* Bx_n = wih + ((size_t)(d * NT_ + 128 + ct) * 16) * 512 + lane * 16;

  v8f c_r[4], c_z[4], c_nh[4], c_nx[4];
#pragma unroll
  for (int m = 0; m < 4; ++m) {
    c_r[m]  = (v8f){0.f, 0.f, 0.f, 0.f, 0.f, 0.f, 0.f, 0.f};
    c_z[m]  = c_r[m];
    c_nh[m] = c_r[m];
    c_nx[m] = c_r[m];
  }

  // Recurrent part: h_prev @ w_hh^T  (K = 1024)
  for (int kk = 0; kk < 32; ++kk) {
    v16bf br = *(const v16bf*)(Bh_r + kk * 512);
    v16bf bz = *(const v16bf*)(Bh_z + kk * 512);
    v16bf bn = *(const v16bf*)(Bh_n + kk * 512);
#pragma unroll
    for (int m = 0; m < 4; ++m) {
      v16bf a = load_a_bf(hrow[m] + kk * 32, half);
      c_r[m]  = WMMA_BF16(a, br, c_r[m]);
      c_z[m]  = WMMA_BF16(a, bz, c_z[m]);
      c_nh[m] = WMMA_BF16(a, bn, c_nh[m]);
    }
  }
  // Input part: x_t @ w_ih^T (K = 512); r,z merge into same accumulators,
  // n-gate x-part kept separate (n = tanh(xn + r*hn)).
  for (int kk = 0; kk < 16; ++kk) {
    v16bf br = *(const v16bf*)(Bx_r + kk * 512);
    v16bf bz = *(const v16bf*)(Bx_z + kk * 512);
    v16bf bn = *(const v16bf*)(Bx_n + kk * 512);
#pragma unroll
    for (int m = 0; m < 4; ++m) {
      v16bf a = load_a_bf(xrow[m] + kk * 32, half);
      c_r[m]  = WMMA_BF16(a, br, c_r[m]);
      c_z[m]  = WMMA_BF16(a, bz, c_z[m]);
      c_nx[m] = WMMA_BF16(a, bn, c_nx[m]);
    }
  }

  const float* bih = d ? b_ih_r : b_ih_f;
  const float* bhh = d ? b_hh_r : b_hh_f;
  const int col = ct * 16 + nIdx;

  const float bias_r  = bih[col]      + bhh[col];
  const float bias_z  = bih[H_ + col] + bhh[H_ + col];
  const float bias_nh = bhh[2 * H_ + col];
  const float bias_nx = bih[2 * H_ + col];

#pragma unroll
  for (int m = 0; m < 4; ++m) {
    for (int r = 0; r < 8; ++r) {
      int b = m * 16 + half * 8 + r;          // C/D layout: VGPR r -> row r + half*8
      float pr = c_r[m][r]  + bias_r;
      float pz = c_z[m][r]  + bias_z;
      float hn = c_nh[m][r] + bias_nh;
      float xn = c_nx[m][r] + bias_nx;
      float hp = h_prev_f[(size_t)b * H_ + col];
      float rg = 1.0f / (1.0f + __expf(-pr));
      float zg = 1.0f / (1.0f + __expf(-pz));
      float ng = tanhf(xn + rg * hn);
      float h  = (1.0f - zg) * ng + zg * hp;
      h_next_f[(size_t)b * H_ + col] = h;
      h_next_bf[(size_t)b * H_ + col] = f2bf(h);
      int opos = t;
      if (d == 1) {
        int len = x_lengths[b];
        opos = (S_ - 1) - ((t + (S_ - len)) & (S_ - 1));  // realign is an involution
      }
      out[((size_t)b * S_ + opos) * (2 * H_) + d * H_ + col] = h;
    }
  }
}

// seq_hidd[b] = hidd_out[b, len[b]-1, :]
__global__ __launch_bounds__(256) void seq_hidd_kernel(
    const int* __restrict__ x_lengths, float* __restrict__ out) {
  int i = blockIdx.x * 256 + threadIdx.x;     // 64 * 2048 = 131,072
  int b = i >> 11;
  int c = i & 2047;
  int len = x_lengths[b];
  out[(size_t)B_ * S_ * 2 * H_ + i] = out[((size_t)b * S_ + (len - 1)) * (2 * H_) + c];
}

extern "C" void kernel_launch(void* const* d_in, const int* in_sizes, int n_in,
                              void* d_out, int out_size, void* d_ws, size_t ws_size,
                              hipStream_t stream) {
  const int*   x_source  = (const int*)d_in[0];
  const int*   x_lengths = (const int*)d_in[1];
  const float* emb       = (const float*)d_in[2];
  const float* w_ih_f    = (const float*)d_in[3];
  const float* w_hh_f    = (const float*)d_in[4];
  const float* b_ih_f    = (const float*)d_in[5];
  const float* b_hh_f    = (const float*)d_in[6];
  const float* w_ih_r    = (const float*)d_in[7];
  const float* w_hh_r    = (const float*)d_in[8];
  const float* b_ih_r    = (const float*)d_in[9];
  const float* b_hh_r    = (const float*)d_in[10];
  float* out = (float*)d_out;

  // Workspace layout (~54 MB total):
  char* ws = (char*)d_ws;
  bf16_t* wih_pack = (bf16_t*)ws;                               //  6,291,456 B
  bf16_t* whh_pack = (bf16_t*)(ws + 6291456);                   // 12,582,912 B
  float*  state_f  = (float*) (ws + 18874368);                  //  1,048,576 B (f32 ping-pong h)
  bf16_t* state_bf = (bf16_t*)(ws + 19922944);                  //    524,288 B (bf16 ping-pong h)
  bf16_t* x_bf     = (bf16_t*)(ws + 20447232);                  // 33,554,432 B (step-ordered inputs)

  pack_wih_kernel<<<12288, 256, 0, stream>>>(w_ih_f, w_ih_r, wih_pack);
  pack_whh_kernel<<<24576, 256, 0, stream>>>(w_hh_f, w_hh_r, whh_pack);
  gather_x_kernel<<<65536, 256, 0, stream>>>(x_source, x_lengths, emb, x_bf);
  zero_state_kernel<<<1024, 256, 0, stream>>>(state_f, state_bf);

  for (int t = 0; t < S_; ++t) {
    gru_step_kernel<<<16, 256, 0, stream>>>(x_lengths, x_bf,
        wih_pack, whh_pack, b_ih_f, b_hh_f, b_ih_r, b_hh_r,
        state_f, state_bf, out, t);
  }

  seq_hidd_kernel<<<512, 256, 0, stream>>>(x_lengths, out);
}
